// MultiHeadAttentionWithLoRA_4157528343230
// MI455X (gfx1250) — matrix operand order repaired and checked by reading, not compiled, not run on hardware
//
#include <hip/hip_runtime.h>

// Problem constants
#define BSZ 4
#define TT 2048
#define CC 1024
#define HH 16
#define HDIM 64
#define RR 8
#define MM 8192          // B*T
#define NQKV 3072        // 3*C
#define SCAL 2.0f

typedef __attribute__((ext_vector_type(16))) __bf16 v16bf;
typedef __attribute__((ext_vector_type(8)))  float  v8f;
typedef __attribute__((ext_vector_type(4)))  unsigned int u32x4;

union FragU { u32x4 q[2]; v16bf v; };

// A-fragment (16x32 bf16): lane holds row (lane&15); hi=lane>>4 selects K-half.
static __device__ __forceinline__ v16bf frag_a16(const __bf16* p) {
  FragU f;
  f.q[0] = *(const u32x4*)(p);
  f.q[1] = *(const u32x4*)(p + 16);
  return f.v;
}
// B-fragment (32x16 bf16) from [n][k] layout: lane holds col (lane&15).
static __device__ __forceinline__ v16bf frag_b16(const __bf16* p) {
  FragU f;
  f.q[0] = *(const u32x4*)(p);
  f.q[1] = *(const u32x4*)(p + 16);
  return f.v;
}
static __device__ __forceinline__ v8f wmma_bf16(v16bf a, v16bf b, v8f c) {
  return __builtin_amdgcn_wmma_f32_16x16x32_bf16(false, a, false, b, (short)0, c,
                                                 false, false);
}
// Async direct global->LDS 16B copy (per lane), tracked by ASYNCcnt.
static __device__ __forceinline__ void async_b128(const __bf16* gsrc, __bf16* ldst) {
  unsigned lo = (unsigned)(unsigned long long)(void*)ldst;          // LDS offset
  unsigned long long ga = (unsigned long long)(const void*)gsrc;    // global addr
  asm volatile("global_load_async_to_lds_b128 %0, %1, off"
               :: "v"(lo), "v"(ga) : "memory");
}
static __device__ __forceinline__ void wait_async0() {
  asm volatile("s_wait_asynccnt 0" ::: "memory");
}

// ---------------- prep kernels ----------------
__global__ void k_cvt_x(const float* __restrict__ x, __bf16* __restrict__ xb, int n) {
  int i = blockIdx.x * blockDim.x + threadIdx.x;
  if (i < n) xb[i] = (__bf16)x[i];
}

// Fold LoRA into dense weights, write transposed [n][k] bf16 (n = p*1024 + h*64 + d)
__global__ void k_build_wqkv(const float* __restrict__ Wq, const float* __restrict__ Wk,
                             const float* __restrict__ Wv, const float* __restrict__ Aq,
                             const float* __restrict__ Bq, const float* __restrict__ Ak,
                             const float* __restrict__ Bk, const float* __restrict__ Av,
                             const float* __restrict__ Bv, __bf16* __restrict__ wt) {
  int idx = blockIdx.x * blockDim.x + threadIdx.x;   // n*1024 + c
  if (idx >= NQKV * CC) return;
  int n = idx >> 10, c = idx & 1023;
  int p = n >> 10, rem = n & 1023, h = rem >> 6, d = rem & 63;
  const float *W, *A, *Bm;
  if (p == 0)      { W = Wq; A = Aq; Bm = Bq; }
  else if (p == 1) { W = Wk; A = Ak; Bm = Bk; }
  else             { W = Wv; A = Av; Bm = Bv; }
  float base = W[((size_t)h * CC + c) * HDIM + d];
  float s = 0.f;
#pragma unroll
  for (int r = 0; r < RR; ++r)
    s += A[((size_t)h * CC + c) * RR + r] * Bm[((size_t)h * RR + r) * HDIM + d];
  wt[(size_t)n * CC + c] = (__bf16)(base + SCAL * s);
}

__global__ void k_build_wp(const float* __restrict__ Wp, const float* __restrict__ Ap,
                           const float* __restrict__ Bp, __bf16* __restrict__ wt) {
  int idx = blockIdx.x * blockDim.x + threadIdx.x;   // n*1024 + c
  if (idx >= CC * CC) return;
  int n = idx >> 10, c = idx & 1023;
  float base = Wp[(size_t)c * CC + n];
  float s = 0.f;
#pragma unroll
  for (int r = 0; r < RR; ++r)
    s += Ap[(size_t)c * RR + r] * Bp[(size_t)r * CC + n];
  wt[(size_t)n * CC + c] = (__bf16)(base + SCAL * s);
}

// ------- WMMA GEMM: 256x128x32 block tile, 8 waves (4M x 2N), 4x4 frags/wave ----
#define GBM 256
#define GBN 128
#define GBK 32
#define LSTR 40   // LDS row stride (elems), multiple of 8 for b128 alignment

__global__ __launch_bounds__(256) void k_gemm(
    const __bf16* __restrict__ A, const __bf16* __restrict__ Bt,  // Bt: [N][K]
    int K, int N, int mode,
    __bf16* __restrict__ qb, __bf16* __restrict__ kb, __bf16* __restrict__ vtb,
    float* __restrict__ outF, const float* __restrict__ bias) {
  __shared__ __bf16 lA[GBM * LSTR];   // 20 KB
  __shared__ __bf16 lB[GBN * LSTR];   // 10 KB
  const int tid = threadIdx.x;
  const int lane = tid & 31;
  const int l15 = lane & 15, hi = lane >> 4;
  const int w = tid >> 5, wm = w >> 1, wn = w & 1;
  const int m0 = blockIdx.y * GBM, n0 = blockIdx.x * GBN;

  v8f zero = {0.f, 0.f, 0.f, 0.f, 0.f, 0.f, 0.f, 0.f};
  v8f acc[4][4];
#pragma unroll
  for (int mt = 0; mt < 4; ++mt)
#pragma unroll
    for (int nt = 0; nt < 4; ++nt) acc[mt][nt] = zero;

  for (int k0 = 0; k0 < K; k0 += GBK) {
    __syncthreads();
    // A tile: 256 rows x 32 cols = 1024 16B chunks, async global->LDS
#pragma unroll
    for (int it = 0; it < 4; ++it) {
      int c = tid + it * 256;
      int r = c >> 2, ch = c & 3;
      async_b128(&A[(size_t)(m0 + r) * K + k0 + ch * 8], &lA[r * LSTR + ch * 8]);
    }
    // B tile: 128 rows x 32 cols = 512 chunks
#pragma unroll
    for (int it = 0; it < 2; ++it) {
      int c = tid + it * 256;
      int r = c >> 2, ch = c & 3;
      async_b128(&Bt[(size_t)(n0 + r) * K + k0 + ch * 8], &lB[r * LSTR + ch * 8]);
    }
    if (k0 + GBK < K) {  // global_prefetch_b8 for next tiles
      int r = tid >> 2, ch = tid & 3;
      __builtin_prefetch(&A[(size_t)(m0 + r) * K + k0 + GBK + ch * 8], 0, 1);
      __builtin_prefetch(&Bt[(size_t)(n0 + (r & 127)) * K + k0 + GBK + ch * 8], 0, 1);
    }
    wait_async0();
    __syncthreads();

    v16bf af[4], bfr[4];
#pragma unroll
    for (int mt = 0; mt < 4; ++mt)
      af[mt] = frag_a16(&lA[(wm * 64 + mt * 16 + l15) * LSTR + hi * 8]);
#pragma unroll
    for (int nt = 0; nt < 4; ++nt)
      bfr[nt] = frag_b16(&lB[(wn * 64 + nt * 16 + l15) * LSTR + hi * 8]);
#pragma unroll
    for (int mt = 0; mt < 4; ++mt)
#pragma unroll
      for (int nt = 0; nt < 4; ++nt)
        acc[mt][nt] = wmma_bf16(af[mt], bfr[nt], acc[mt][nt]);
  }

  // Epilogue. C layout: row = i + 8*hi, col = lane&15.
#pragma unroll
  for (int mt = 0; mt < 4; ++mt) {
#pragma unroll
    for (int nt = 0; nt < 4; ++nt) {
#pragma unroll
      for (int i = 0; i < 8; ++i) {
        int mg = m0 + wm * 64 + mt * 16 + i + 8 * hi;
        int ng = n0 + wn * 64 + nt * 16 + l15;
        float v = acc[mt][nt][i];
        if (mode == 0) {
          int p = ng >> 10, rem = ng & 1023, h = rem >> 6, d = rem & 63;
          int bb = mg >> 11, t = mg & 2047;
          size_t bhh = (size_t)(bb * HH + h);
          if (p == 0)      qb[(bhh * TT + t) * HDIM + d] = (__bf16)(v * 0.125f);
          else if (p == 1) kb[(bhh * TT + t) * HDIM + d] = (__bf16)v;
          else             vtb[(bhh * HDIM + d) * TT + t] = (__bf16)v;  // V^T
        } else {
          outF[(size_t)mg * N + ng] = v + bias[ng];
        }
      }
    }
  }
}

// ---------------- flash attention (causal) ----------------
// Block: 128 q rows, 8 waves x 16-row stripes; s-tiles of 64 keys.
#define SBN 64
#define KSTR 72   // K-tile LDS stride
#define VSTR 72   // V^T-tile LDS stride
#define PSTR 72   // P stripe stride

__global__ __launch_bounds__(256) void k_attn(const __bf16* __restrict__ qb,
                                              const __bf16* __restrict__ kb,
                                              const __bf16* __restrict__ vtb,
                                              __bf16* __restrict__ ao) {
  extern __shared__ __bf16 smem[];
  __bf16* kt = smem;                     // 64 x KSTR
  __bf16* vt = kt + SBN * KSTR;          // 64 x VSTR  (rows = d, cols = s)
  __bf16* pb = vt + HDIM * VSTR;         // 8 waves x 16 x PSTR

  const int tid = threadIdx.x;
  const int lane = tid & 31;
  const int l15 = lane & 15, hi = lane >> 4;
  const int w = tid >> 5;
  const int bh = blockIdx.y;             // b*16 + h
  const int q0 = blockIdx.x * 128;
  const int r0 = q0 + w * 16;            // this wave's query-row base
  const size_t qbase = (size_t)bh * TT * HDIM;

  // Q fragments (rows r0.., K = head dim, 2 k-steps of 32), softmax scale pre-folded
  v16bf qf[2];
#pragma unroll
  for (int ks = 0; ks < 2; ++ks)
    qf[ks] = frag_a16(&qb[qbase + (size_t)(r0 + l15) * HDIM + ks * 32 + hi * 8]);

  v8f zero = {0.f, 0.f, 0.f, 0.f, 0.f, 0.f, 0.f, 0.f};
  v8f o[4];
  float mrow[8], lrow[8];
#pragma unroll
  for (int dt = 0; dt < 4; ++dt) o[dt] = zero;
#pragma unroll
  for (int i = 0; i < 8; ++i) { mrow[i] = -1e30f; lrow[i] = 0.f; }

  __bf16* pw = pb + w * 16 * PSTR;
  const int nst = 2 * (blockIdx.x + 1);  // s-tiles needed (causal)

  for (int st = 0; st < nst; ++st) {
    const int s0 = st * SBN;
    __syncthreads();
    // Stage K tile [s][d] and V^T tile [d][s] via async global->LDS
#pragma unroll
    for (int it = 0; it < 2; ++it) {
      int c = tid + it * 256;
      int r = c >> 3, ch = c & 7;        // 64 rows x 8 chunks
      async_b128(&kb[qbase + (size_t)(s0 + r) * HDIM + ch * 8],
                 &kt[r * KSTR + ch * 8]);
      async_b128(&vtb[((size_t)bh * HDIM + r) * TT + s0 + ch * 8],
                 &vt[r * VSTR + ch * 8]);
    }
    wait_async0();
    __syncthreads();

    if (s0 <= r0 + 15) {                 // tile not fully above diagonal for this wave
      // S = Q K^T
      v8f sacc[4];
#pragma unroll
      for (int nt = 0; nt < 4; ++nt) {
        sacc[nt] = zero;
#pragma unroll
        for (int ks = 0; ks < 2; ++ks) {
          v16bf bfg = frag_b16(&kt[(nt * 16 + l15) * KSTR + ks * 32 + hi * 8]);
          sacc[nt] = wmma_bf16(qf[ks], bfg, sacc[nt]);
        }
      }
      // Causal mask on partial tiles
      if (s0 + SBN - 1 > r0) {
#pragma unroll
        for (int nt = 0; nt < 4; ++nt)
#pragma unroll
          for (int i = 0; i < 8; ++i) {
            int col = s0 + nt * 16 + l15;
            int row = r0 + i + 8 * hi;
            if (col > row) sacc[nt][i] = -1e30f;
          }
      }
      // Online softmax: row = i + 8*hi lives on 16 lanes of one half-wave
#pragma unroll
      for (int i = 0; i < 8; ++i) {
        float mx = -1e30f;
#pragma unroll
        for (int nt = 0; nt < 4; ++nt) mx = fmaxf(mx, sacc[nt][i]);
#pragma unroll
        for (int msk = 1; msk < 16; msk <<= 1)
          mx = fmaxf(mx, __shfl_xor(mx, msk, 32));
        float mnew = fmaxf(mrow[i], mx);
        float corr = __expf(mrow[i] - mnew);
        mrow[i] = mnew;
        float rs = 0.f;
#pragma unroll
        for (int nt = 0; nt < 4; ++nt) {
          float pv = __expf(sacc[nt][i] - mnew);
          sacc[nt][i] = pv;
          rs += pv;
        }
#pragma unroll
        for (int msk = 1; msk < 16; msk <<= 1) rs += __shfl_xor(rs, msk, 32);
        lrow[i] = lrow[i] * corr + rs;
#pragma unroll
        for (int dt = 0; dt < 4; ++dt) o[dt][i] *= corr;
      }
      // C-layout -> A-layout via per-wave LDS stripe
#pragma unroll
      for (int nt = 0; nt < 4; ++nt)
#pragma unroll
        for (int i = 0; i < 8; ++i)
          pw[(i + 8 * hi) * PSTR + nt * 16 + l15] = (__bf16)sacc[nt][i];
      asm volatile("s_wait_dscnt 0" ::: "memory");
      // O += P V
#pragma unroll
      for (int ks = 0; ks < 2; ++ks) {
        v16bf af = frag_a16(&pw[l15 * PSTR + ks * 32 + hi * 8]);
#pragma unroll
        for (int dt = 0; dt < 4; ++dt) {
          v16bf bfg = frag_b16(&vt[(dt * 16 + l15) * VSTR + ks * 32 + hi * 8]);
          o[dt] = wmma_bf16(af, bfg, o[dt]);
        }
      }
    }
  }

  // Write attention output as bf16 [b, t, h*64+d] for the final projection GEMM
  const int b = bh >> 4, h = bh & 15;
#pragma unroll
  for (int dt = 0; dt < 4; ++dt)
#pragma unroll
    for (int i = 0; i < 8; ++i) {
      int t = r0 + i + 8 * hi;
      float v = o[dt][i] / lrow[i];
      ao[((size_t)(b * TT + t)) * CC + h * HDIM + dt * 16 + l15] = (__bf16)v;
    }
}

// ---------------- launcher ----------------
extern "C" void kernel_launch(void* const* d_in, const int* in_sizes, int n_in,
                              void* d_out, int out_size, void* d_ws, size_t ws_size,
                              hipStream_t stream) {
  (void)in_sizes; (void)n_in; (void)out_size; (void)ws_size;
  const float* x  = (const float*)d_in[0];
  const float* Wq = (const float*)d_in[1];
  const float* Wk = (const float*)d_in[2];
  const float* Wv = (const float*)d_in[3];
  const float* Aq = (const float*)d_in[4];
  const float* Bq = (const float*)d_in[5];
  const float* Ak = (const float*)d_in[6];
  const float* Bk = (const float*)d_in[7];
  const float* Av = (const float*)d_in[8];
  const float* Bv = (const float*)d_in[9];
  const float* Wp = (const float*)d_in[10];
  const float* bp = (const float*)d_in[11];
  const float* Ap = (const float*)d_in[12];
  const float* Bp = (const float*)d_in[13];
  float* out = (float*)d_out;

  char* ws = (char*)d_ws;
  size_t off = 0;
  __bf16* xb   = (__bf16*)(ws + off); off += (size_t)MM * CC * 2;        // 16 MB
  __bf16* wqkv = (__bf16*)(ws + off); off += (size_t)NQKV * CC * 2;      // 6 MB
  __bf16* wpt  = (__bf16*)(ws + off); off += (size_t)CC * CC * 2;        // 2 MB
  __bf16* qb   = (__bf16*)(ws + off); off += (size_t)MM * CC * 2;        // 16 MB
  __bf16* kbuf = (__bf16*)(ws + off); off += (size_t)MM * CC * 2;        // 16 MB
  __bf16* vtb  = (__bf16*)(ws + off); off += (size_t)MM * CC * 2;        // 16 MB
  __bf16* ao   = (__bf16*)(ws + off); off += (size_t)MM * CC * 2;        // 16 MB

  k_cvt_x<<<(MM * CC) / 256, 256, 0, stream>>>(x, xb, MM * CC);
  k_build_wqkv<<<(NQKV * CC) / 256, 256, 0, stream>>>(Wq, Wk, Wv, Aq, Bq, Ak, Bk,
                                                      Av, Bv, wqkv);
  k_build_wp<<<(CC * CC) / 256, 256, 0, stream>>>(Wp, Ap, Bp, wpt);

  // QKV projection: [8192,1024] x [1024,3072]
  k_gemm<<<dim3(NQKV / GBN, MM / GBM), 256, 0, stream>>>(
      xb, wqkv, CC, NQKV, 0, qb, kbuf, vtb, nullptr, nullptr);

  // Flash attention
  size_t smem = (size_t)(SBN * KSTR + HDIM * VSTR + 8 * 16 * PSTR) * 2;
  k_attn<<<dim3(TT / 128, BSZ * HH), 256, smem, stream>>>(qb, kbuf, vtb, ao);

  // Output projection + bias: [8192,1024] x [1024,1024]
  k_gemm<<<dim3(CC / GBN, MM / GBM), 256, 0, stream>>>(
      ao, wpt, CC, CC, 1, nullptr, nullptr, nullptr, out, bp);
}
